// EgnnMixPolicy_38448547234260
// MI455X (gfx1250) — compile-verified, matrix operand order checked by reading, not run on hardware
//
#include <hip/hip_runtime.h>
#include <hip/hip_bf16.h>
#include <math.h>

#define HID 128
#define NNODES 64
#define NB_EDGES 4032            // 63*64 edges per graph
#define ET 16                    // edges per tile (one 16-row WMMA M-tile)
#define EDGE_ITERS (NB_EDGES/ET) // 252

// Fragment table in d_ws: each frag = 32 lanes x 16 halves = 512 halves = 1KB.
#define OFF_EMBED 0     // 8 frags  (K=32 pad of 30)
#define OFF_VELM1 8     // 32 frags (K=128)
#define OFF_MLP1  40    // 16 frags (K=64 pad of 34)
#define OFF_MLP2  56    // 32 frags
#define OFF_E1A   88    // 32 frags (edge_w1 rows 0..127)
#define OFF_E1B   120   // 32 frags (edge_w1 rows 128..255)
#define OFF_E2    152   // 32 frags (edge_w2)
#define OFF_C1    184   // 32 frags (coord_w1)
#define N_FRAGS   216

typedef __attribute__((ext_vector_type(16))) _Float16 v16h;
typedef __attribute__((ext_vector_type(8)))  float    v8f;

struct EgnnParams {
  const float* obs; const float* eps;
  const float* embed_b;
  const float* edge_b1; const float* edge_b2;
  const float* coord_b1; const float* coord_w2; const float* coord_b2;
  const float* velm_b1; const float* velm_w2; const float* velm_b2;
  const float* mlp_b1; const float* mlp_b2;
  const float* mlp_w3; const float* mlp_b3;
  const float* log_std;
  const float* edge_w1;           // for the radial rows (256,257)
  const _Float16* ws;             // packed B fragments
  float* out;
};

// K index inside a 16-half fragment per the CDNA5 16-bit A/B layout:
// VGPR j holds K = koff+2j,koff+2j+1 (j<4) and 16+... (j>=4); koff=8 for lanes 16-31.
__device__ __forceinline__ int kmap16(int h, int lane) {
  int koff = (lane & 16) ? 8 : 0;
  return (h < 8) ? (koff + h) : (16 + koff + (h - 8));
}

// A fragment (16x32 f16) from LDS, row-major, leading dim `ld` halves.
__device__ __forceinline__ v16h ldsA16(const _Float16* S, int ld, int mbase, int kbase, int lane) {
  v16h a;
  const _Float16* row = S + (mbase + (lane & 15)) * ld + kbase;
#pragma unroll
  for (int h = 0; h < 16; ++h) a[h] = row[kmap16(h, lane)];
  return a;
}

// B fragment from pre-packed workspace: one aligned 32-byte load per lane.
__device__ __forceinline__ v16h wsB(const _Float16* ws, int frag, int lane) {
  return *(const v16h*)(ws + (size_t)frag * 512 + lane * 16);
}

__device__ __forceinline__ float silu(float x) { return x / (1.0f + __expf(-x)); }

// Store a 16x16 f32 accumulator tile (+bias, optional silu) to LDS as f16.
// C layout: VGPR i, lanes 0-15 -> M = i ; lanes 16-31 -> M = 8+i ; N = lane&15.
__device__ __forceinline__ void storeC16(_Float16* S, int ld, int mbase, int nbase,
                                         v8f acc, float bias, bool do_silu, int lane) {
  int n  = nbase + (lane & 15);
  int mo = mbase + 8 * (lane >> 4);
#pragma unroll
  for (int i = 0; i < 8; ++i) {
    float x = acc[i] + bias;
    if (do_silu) x = silu(x);
    S[(mo + i) * ld + n] = (_Float16)x;
  }
}

#define WMMA_F16(a, b, c) \
  __builtin_amdgcn_wmma_f32_16x16x32_f16(false, (a), false, (b), (short)0, (c), false, false)

// ---- prep: pack a K x 128 f32 weight matrix into B-fragment-layout f16 ----
// grid.x = 8 ntiles * ksteps ; 512 threads, one per fragment element.
__global__ __launch_bounds__(512) void pack_weights_kernel(const float* __restrict__ W,
                                                           _Float16* __restrict__ dst,
                                                           int ksteps, int kmax) {
  int frag  = blockIdx.x;
  int ntile = frag / ksteps;
  int ks    = frag - ntile * ksteps;
  int tid   = threadIdx.x;
  int lane  = tid >> 4;
  int h     = tid & 15;
  int n = ntile * 16 + (lane & 15);
  int k = ks * 32 + kmap16(h, lane);
  float v = (k < kmax) ? W[(size_t)k * HID + n] : 0.0f;
  dst[(size_t)frag * 512 + lane * 16 + h] = (_Float16)v;
}

__global__ __launch_bounds__(256) void egnn_policy_kernel(EgnnParams p) {
  __shared__ _Float16 sH[NNODES * HID];   // node embedding h (f16)
  __shared__ _Float16 sP[NNODES * HID];   // H@W1a + b1   (also velm-s / z2 staging)
  __shared__ _Float16 sQ[NNODES * HID];   // H@W1b        (also z1 staging)
  __shared__ _Float16 sM[2 * ET * HID];   // edge tiles m1|m2 (also h_in / z staging)
  __shared__ float sLoc[NNODES][2], sVel[NNODES][2];
  __shared__ float sWrc[HID];             // edge_w1 rows 256+257 combined
  __shared__ float sCwP[8][ET];           // per-wave cw partials (no atomics)
  __shared__ float sAggX[NNODES][2];
  __shared__ float sVelScale[NNODES];
  __shared__ float sMuD[NNODES][2];

  const int tid   = threadIdx.x;
  const int lane  = tid & 31;
  const int wv    = tid >> 5;      // 8 waves
  const int nbase = wv * 16;       // this wave's 16 output columns
  const int g     = blockIdx.x;    // graph id (128 graphs)
  const int n0    = g * NNODES;
  const _Float16* ws = p.ws;

  _Float16* sM1 = sM;
  _Float16* sM2 = sM + ET * HID;

  // ---------- phase 0: stage obs ----------
  for (int i = tid; i < NNODES; i += 256) {
    const float* o = p.obs + (size_t)(n0 + i) * 34;
    sLoc[i][0] = o[0]; sLoc[i][1] = o[1];
    sVel[i][0] = o[2]; sVel[i][1] = o[3];
    sAggX[i][0] = 0.f; sAggX[i][1] = 0.f;
  }
  if (tid < HID) sWrc[tid] = p.edge_w1[256 * HID + tid] + p.edge_w1[257 * HID + tid];
  // h_in (64x30 -> f16 64x32, K padded) into sM scratch
  for (int idx = tid; idx < NNODES * 32; idx += 256) {
    int i = idx >> 5, k = idx & 31;
    sM[idx] = (k < 30) ? (_Float16)p.obs[(size_t)(n0 + i) * 34 + 4 + k] : (_Float16)0.f;
  }
  __syncthreads();

  // ---------- phase 1: h = h_in @ embed_w + b (K=32, single WMMA per tile) ----------
  {
    float bias = p.embed_b[nbase + (lane & 15)];
    v16h b = wsB(ws, OFF_EMBED + wv, lane);
#pragma unroll
    for (int mt = 0; mt < 4; ++mt) {
      v8f acc = {};
      v16h a = ldsA16(sM, 32, mt * 16, 0, lane);
      acc = WMMA_F16(a, b, acc);
      storeC16(sH, HID, mt * 16, nbase, acc, bias, false, lane);
    }
  }
  __syncthreads();

  // ---------- phase 2: velm head: s = silu(h @ velm_w1 + b1) -> sP ----------
  {
    float bias = p.velm_b1[nbase + (lane & 15)];
    for (int mt = 0; mt < 4; ++mt) {
      v8f acc = {};
#pragma unroll
      for (int ks = 0; ks < 4; ++ks) {
        v16h a = ldsA16(sH, HID, mt * 16, ks * 32, lane);
        acc = WMMA_F16(a, wsB(ws, OFF_VELM1 + wv * 4 + ks, lane), acc);
      }
      storeC16(sP, HID, mt * 16, nbase, acc, bias, true, lane);
    }
  }
  __syncthreads();
  if (tid < NNODES) {  // vel_scale = s @ velm_w2 + b2
    float acc = p.velm_b2[0];
    for (int k = 0; k < HID; ++k) acc += (float)sP[tid * HID + k] * p.velm_w2[k];
    sVelScale[tid] = acc;
  }
  __syncthreads();

  // ---------- phase 3: obs MLP (z 64x34, K padded to 64) ----------
  for (int idx = tid; idx < NNODES * 64; idx += 256) {  // build z f16 into sM
    int i = idx >> 6, k = idx & 63;
    float v;
    if (k < 2)       v = sLoc[i][k];
    else if (k < 4)  v = sVel[i][k - 2];
    else if (k < 34) v = p.obs[(size_t)(n0 + i) * 34 + 4 + (k - 4)];
    else             v = 0.f;
    sM[idx] = (_Float16)v;
  }
  __syncthreads();
  {  // z1 = silu(z @ mlp_w1 + b1) -> sQ
    float bias = p.mlp_b1[nbase + (lane & 15)];
    for (int mt = 0; mt < 4; ++mt) {
      v8f acc = {};
#pragma unroll
      for (int ks = 0; ks < 2; ++ks) {
        v16h a = ldsA16(sM, 64, mt * 16, ks * 32, lane);
        acc = WMMA_F16(a, wsB(ws, OFF_MLP1 + wv * 2 + ks, lane), acc);
      }
      storeC16(sQ, HID, mt * 16, nbase, acc, bias, true, lane);
    }
  }
  __syncthreads();
  {  // z2 = silu(z1 @ mlp_w2 + b2) -> sP
    float bias = p.mlp_b2[nbase + (lane & 15)];
    for (int mt = 0; mt < 4; ++mt) {
      v8f acc = {};
#pragma unroll
      for (int ks = 0; ks < 4; ++ks) {
        v16h a = ldsA16(sQ, HID, mt * 16, ks * 32, lane);
        acc = WMMA_F16(a, wsB(ws, OFF_MLP2 + wv * 4 + ks, lane), acc);
      }
      storeC16(sP, HID, mt * 16, nbase, acc, bias, true, lane);
    }
  }
  __syncthreads();
  if (tid < 128) {  // mu_delta = z2 @ mlp_w3 + b3  (N=2, VALU dot)
    int i = tid >> 1, j = tid & 1;
    float acc = p.mlp_b3[j];
    for (int k = 0; k < HID; ++k) acc += (float)sP[i * HID + k] * p.mlp_w3[k * 2 + j];
    sMuD[i][j] = acc;
  }
  __syncthreads();

  // ---------- phase 4: P = h@W1a + b1, Q = h@W1b (edge layer-1 decomposition) ----------
  {
    float bias = p.edge_b1[nbase + (lane & 15)];
    for (int mt = 0; mt < 4; ++mt) {
      v8f accP = {}, accQ = {};
#pragma unroll
      for (int ks = 0; ks < 4; ++ks) {
        v16h a = ldsA16(sH, HID, mt * 16, ks * 32, lane);
        accP = WMMA_F16(a, wsB(ws, OFF_E1A + wv * 4 + ks, lane), accP);
        accQ = WMMA_F16(a, wsB(ws, OFF_E1B + wv * 4 + ks, lane), accQ);
      }
      storeC16(sP, HID, mt * 16, nbase, accP, bias, false, lane);
      storeC16(sQ, HID, mt * 16, nbase, accQ, 0.0f, false, lane);
    }
  }

  // ---------- preload edge-loop weights into registers (per-wave column strip) ----------
  v16h Bw2[4], Bc1[4];
#pragma unroll
  for (int ks = 0; ks < 4; ++ks) {
    Bw2[ks] = wsB(ws, OFF_E2 + wv * 4 + ks, lane);
    Bc1[ks] = wsB(ws, OFF_C1 + wv * 4 + ks, lane);
  }
  const float b2n  = p.edge_b2[nbase + (lane & 15)];
  const float bc1n = p.coord_b1[nbase + (lane & 15)];
  const float wc2n = p.coord_w2[nbase + (lane & 15)];
  const float cb2  = p.coord_b2[0];
  __syncthreads();

  // ---------- phase 5: edge loop (252 tiles of 16 edges) ----------
  for (int it = 0; it < EDGE_ITERS; ++it) {
    const int e0 = it * ET;
    {  // assemble m1 = silu(P[r] + Q[c] + radial*w_rc): 16 threads per edge, 8 cols each
      int eIdx = tid >> 4;
      int nb   = (tid & 15) * 8;
      int q = e0 + eIdx;
      int r = q / 63;
      int kk = q - r * 63;
      int c = kk + (kk >= r);
      float dx = sLoc[r][0] - sLoc[c][0], dy = sLoc[r][1] - sLoc[c][1];
      float radial = dx * dx + dy * dy;
#pragma unroll
      for (int jj = 0; jj < 8; ++jj) {
        int n = nb + jj;
        float v = (float)sP[r * HID + n] + (float)sQ[c * HID + n] + radial * sWrc[n];
        sM1[eIdx * HID + n] = (_Float16)silu(v);
      }
    }
    __syncthreads();
    {  // m2 = silu(m1 @ edge_w2 + b2)
      v8f acc = {};
#pragma unroll
      for (int ks = 0; ks < 4; ++ks) {
        v16h a = ldsA16(sM1, HID, 0, ks * 32, lane);
        acc = WMMA_F16(a, Bw2[ks], acc);
      }
      storeC16(sM2, HID, 0, nbase, acc, b2n, true, lane);
    }
    __syncthreads();
    {  // t = silu(m2 @ coord_w1 + bc1); cw partial = sum_n t * coord_w2[n]
      v8f acc = {};
#pragma unroll
      for (int ks = 0; ks < 4; ++ks) {
        v16h a = ldsA16(sM2, HID, 0, ks * 32, lane);
        acc = WMMA_F16(a, Bc1[ks], acc);
      }
      // In-wave cross-lane reduction over the 16 columns this wave owns:
      // rows i (lanes 0-15) and 8+i (lanes 16-31); xor masks 1,2,4,8 stay in-half.
#pragma unroll
      for (int i = 0; i < 8; ++i) {
        float v = silu(acc[i] + bc1n) * wc2n;
        v += __shfl_xor(v, 1);
        v += __shfl_xor(v, 2);
        v += __shfl_xor(v, 4);
        v += __shfl_xor(v, 8);
        if ((lane & 15) == 0) sCwP[wv][i + 8 * (lane >> 4)] = v;
      }
    }
    __syncthreads();
    if (tid < ET) {  // agg_x[r] += diff/(sqrt(radial)+1) * cw
      float cw = cb2;
#pragma unroll
      for (int w = 0; w < 8; ++w) cw += sCwP[w][tid];
      int q = e0 + tid;
      int r = q / 63;
      int kk = q - r * 63;
      int c = kk + (kk >= r);
      float dx = sLoc[r][0] - sLoc[c][0], dy = sLoc[r][1] - sLoc[c][1];
      float radial = dx * dx + dy * dy;
      float inv = 1.0f / (sqrtf(radial) + 1.0f);
      atomicAdd(&sAggX[r][0], dx * inv * cw);
      atomicAdd(&sAggX[r][1], dy * inv * cw);
    }
    __syncthreads();
  }

  // ---------- phase 6: outputs ----------
  if (tid < NNODES) {
    int i = tid;
    float vx = sVelScale[i] * sVel[i][0] + sAggX[i][0];
    float vy = sVelScale[i] * sVel[i][1] + sAggX[i][1];
    float mu0 = vx + sMuD[i][0];
    float mu1 = vy + sMuD[i][1];
    float ls0 = p.log_std[0], ls1 = p.log_std[1];
    float s0 = __expf(ls0), s1 = __expf(ls1);
    float e0 = p.eps[(size_t)(n0 + i) * 2 + 0];
    float e1 = p.eps[(size_t)(n0 + i) * 2 + 1];
    float a0 = mu0 + s0 * e0, a1 = mu1 + s1 * e1;
    const float C = 0.91893853320467274178f;  // 0.5*log(2*pi)
    float d0 = (a0 - mu0) / s0, d1 = (a1 - mu1) / s1;
    float* o = p.out + (size_t)(n0 + i) * 4;
    o[0] = a0;
    o[1] = a1;
    o[2] = -0.5f * d0 * d0 - ls0 - C;
    o[3] = -0.5f * d1 * d1 - ls1 - C;
  }
}

extern "C" void kernel_launch(void* const* d_in, const int* in_sizes, int n_in,
                              void* d_out, int out_size, void* d_ws, size_t ws_size,
                              hipStream_t stream) {
  (void)in_sizes; (void)n_in; (void)out_size; (void)ws_size;
  const float* embed_w  = (const float*)d_in[2];
  const float* edge_w1  = (const float*)d_in[4];
  const float* edge_w2  = (const float*)d_in[6];
  const float* coord_w1 = (const float*)d_in[8];
  const float* velm_w1  = (const float*)d_in[12];
  const float* mlp_w1   = (const float*)d_in[20];
  const float* mlp_w2   = (const float*)d_in[22];

  _Float16* ws = (_Float16*)d_ws;  // needs N_FRAGS KB = 216 KB of scratch

  // Pack all B-operand weights into WMMA fragment layout (f16, zero-padded K).
  pack_weights_kernel<<<8,  512, 0, stream>>>(embed_w,             ws + OFF_EMBED * 512, 1, 30);
  pack_weights_kernel<<<32, 512, 0, stream>>>(velm_w1,             ws + OFF_VELM1 * 512, 4, 128);
  pack_weights_kernel<<<16, 512, 0, stream>>>(mlp_w1,              ws + OFF_MLP1  * 512, 2, 34);
  pack_weights_kernel<<<32, 512, 0, stream>>>(mlp_w2,              ws + OFF_MLP2  * 512, 4, 128);
  pack_weights_kernel<<<32, 512, 0, stream>>>(edge_w1,             ws + OFF_E1A   * 512, 4, 128);
  pack_weights_kernel<<<32, 512, 0, stream>>>(edge_w1 + 128 * HID, ws + OFF_E1B   * 512, 4, 128);
  pack_weights_kernel<<<32, 512, 0, stream>>>(edge_w2,             ws + OFF_E2    * 512, 4, 128);
  pack_weights_kernel<<<32, 512, 0, stream>>>(coord_w1,            ws + OFF_C1    * 512, 4, 128);

  EgnnParams p;
  p.obs      = (const float*)d_in[0];
  p.eps      = (const float*)d_in[1];
  p.embed_b  = (const float*)d_in[3];
  p.edge_w1  = edge_w1;
  p.edge_b1  = (const float*)d_in[5];
  p.edge_b2  = (const float*)d_in[7];
  p.coord_b1 = (const float*)d_in[9];
  p.coord_w2 = (const float*)d_in[10];
  p.coord_b2 = (const float*)d_in[11];
  p.velm_b1  = (const float*)d_in[13];
  p.velm_w2  = (const float*)d_in[14];
  p.velm_b2  = (const float*)d_in[15];
  // d_in[16..19] = node_w1/b1/w2/b2 : dead code w.r.t. the reference output
  p.mlp_b1   = (const float*)d_in[21];
  p.mlp_b2   = (const float*)d_in[23];
  p.mlp_w3   = (const float*)d_in[24];
  p.mlp_b3   = (const float*)d_in[25];
  p.log_std  = (const float*)d_in[26];
  // d_in[27], d_in[28] = rows/cols : derived analytically in-kernel
  p.ws       = ws;
  p.out      = (float*)d_out;
  egnn_policy_kernel<<<128, 256, 0, stream>>>(p);
}
